// GINLayer_8160437862943
// MI455X (gfx1250) — compile-verified, matrix operand order checked by reading, not run on hardware
//
#include <hip/hip_runtime.h>
#include <hip/hip_bf16.h>

// ---------------------------------------------------------------------------
// GIN layer for MI455X (gfx1250, wave32, WMMA).
//   agg = segment_sum(node_feats[src] + emb0[e0]+b0 + emb1[e1]+b1, dst)
//   h   = relu(agg @ W1 + b1) @ W2 + b2 ; BatchNorm(training stats)
// GEMMs run on v_wmma_f32_16x16x32_bf16 (f32 accumulate).
//   - one-shot f32->bf16 convert pass keeps conversions out of the GEMM loop
//   - each wave: 16x64 strip (4 WMMA accumulators) -> 4 WMMAs per A fragment
// ---------------------------------------------------------------------------

#define NNODES   100000
#define NEDGES   200000
#define DFEAT    300
#define DHID     600

#define MPAD     100032          // 1563 * 64
#define MTILES   1563
#define K1PAD    320             // D padded,  KT1 = 10
#define N1PAD    640             // 2D padded, 5 column-blocks of 128
#define K2PAD    640             // KT2 = 20
#define N2PAD    384             // 3 column-blocks of 128 (cols >= 300 guarded)
#define KT1      10
#define NT1      40              // N1PAD / 16
#define KT2      20
#define NT2      24              // N2PAD / 16

typedef __attribute__((ext_vector_type(16))) __bf16 v16bf;
typedef __attribute__((ext_vector_type(8)))  float  v8f;

union FragA {
    v16bf          b;
    uint4          q[2];
    unsigned short us[16];
};

__device__ __forceinline__ unsigned short f2bf(float x) {
    unsigned int u = __float_as_uint(x);
    unsigned int r = u + 0x7FFFu + ((u >> 16) & 1u);   // round-to-nearest-even
    return (unsigned short)(r >> 16);
}

// ---------------------------------------------------------------------------
// 0) zero the padded aggregation buffer (must happen every call: atomics)
// ---------------------------------------------------------------------------
__global__ void zero_f32_kernel(float* __restrict__ p, size_t n) {
    size_t i = (size_t)blockIdx.x * blockDim.x + threadIdx.x;
    if (i < n) p[i] = 0.0f;
}

// ---------------------------------------------------------------------------
// 1) edge scatter: agg[dst] += node_feats[src] + emb0[e0]+b0 + emb1[e1]+b1
// ---------------------------------------------------------------------------
__global__ __launch_bounds__(128)
void scatter_kernel(const float* __restrict__ nf,
                    const int* __restrict__ src, const int* __restrict__ dst,
                    const int* __restrict__ e0,  const int* __restrict__ e1,
                    const float* __restrict__ w0, const float* __restrict__ b0,
                    const float* __restrict__ w1, const float* __restrict__ b1,
                    float* __restrict__ agg) {
    int e = blockIdx.x;
    if (e >= NEDGES) return;
    const float* nfr = nf + (size_t)src[e] * DFEAT;
    const float* w0r = w0 + (size_t)e0[e] * DFEAT;
    const float* w1r = w1 + (size_t)e1[e] * DFEAT;
    float* ar = agg + (size_t)dst[e] * K1PAD;
    for (int i = threadIdx.x; i < DFEAT; i += blockDim.x) {
        float v = nfr[i] + w0r[i] + b0[i] + w1r[i] + b1[i];
        atomicAdd(ar + i, v);
    }
}

// ---------------------------------------------------------------------------
// 2) elementwise f32 -> bf16 (each agg element converted exactly once)
// ---------------------------------------------------------------------------
__global__ void cvt_bf16_kernel(const float* __restrict__ in,
                                unsigned short* __restrict__ o, size_t n) {
    size_t i = (size_t)blockIdx.x * blockDim.x + threadIdx.x;
    if (i < n) o[i] = f2bf(in[i]);
}

// ---------------------------------------------------------------------------
// 3) pack an f32 K x N weight matrix into bf16 WMMA B-fragments.
//    Fragment t = nt*KT + kt holds 32 lanes x 16 bf16.
//    lane<16 : col = nt*16+lane,    K = kt*32 + {i<8 ? i : i+8}
//    lane>=16: col = nt*16+lane-16, K = kt*32 + 8 + {i<8 ? i : i+8}
// ---------------------------------------------------------------------------
__global__ void pack_w_kernel(const float* __restrict__ w,
                              unsigned short* __restrict__ wp,
                              int KT, int NT, int Krows, int Ncols) {
    int idx = blockIdx.x * blockDim.x + threadIdx.x;
    int total = KT * NT * 512;
    if (idx >= total) return;
    int t      = idx >> 9;
    int within = idx & 511;
    int lane   = within >> 4;
    int i      = within & 15;
    int kt = t % KT;
    int nt = t / KT;
    int n = nt * 16 + (lane & 15);
    int k = kt * 32 + (i & 7) + ((i >> 3) << 4) + ((lane >> 4) << 3);
    float v = (k < Krows && n < Ncols) ? w[(size_t)k * Ncols + n] : 0.0f;
    wp[idx] = f2bf(v);
}

// ---------------------------------------------------------------------------
// 4) GEMM: bf16 A (row-major, padded) x prepacked bf16 B, f32 accumulate.
//    256 threads = 8 waves; block tile 64(M) x 128(N); wave strip 16x64.
//    FIRST: out = relu(acc + b1) -> bf16 H (stride N1PAD, all pads written)
//    else : out = acc + b2       -> f32 d_out (guarded to NNODES x DFEAT)
// ---------------------------------------------------------------------------
template<int KT, int ASTRIDE, bool FIRST>
__global__ __launch_bounds__(256)
void gemm_kernel(const unsigned short* __restrict__ A,
                 const unsigned short* __restrict__ Wp,
                 const float* __restrict__ bias,
                 void* __restrict__ outv) {
    const int lane   = threadIdx.x & 31;
    const int wave   = threadIdx.x >> 5;
    const int wm     = wave & 3;
    const int wn     = wave >> 2;
    const int m_base = blockIdx.x * 64 + wm * 16;
    const int row    = m_base + (lane & 15);
    const int khalf  = lane >> 4;
    const int nt0    = blockIdx.y * 8 + wn * 4;

    v8f c[4] = {{}, {}, {}, {}};
    for (int kt = 0; kt < KT; ++kt) {
        const unsigned short* ap = A + (size_t)row * ASTRIDE + kt * 32 + khalf * 8;
        FragA fa;
        fa.q[0] = *(const uint4*)ap;
        fa.q[1] = *(const uint4*)(ap + 16);
#pragma unroll
        for (int s = 0; s < 4; ++s) {
            v16bf b = *(const v16bf*)(Wp + ((size_t)(nt0 + s) * KT + kt) * 512 + lane * 16);
            c[s] = __builtin_amdgcn_wmma_f32_16x16x32_bf16(false, fa.b, false, b,
                                                           (short)0, c[s], false, false);
        }
    }

    // C layout: VGPR i, lane l -> M = i + (l>=16)*8, N = l&15
    const int rbase = m_base + khalf * 8;
#pragma unroll
    for (int s = 0; s < 4; ++s) {
        const int n = (nt0 + s) * 16 + (lane & 15);
        if constexpr (FIRST) {
            unsigned short* H = (unsigned short*)outv;
            float bi = (n < DHID) ? bias[n] : 0.0f;
#pragma unroll
            for (int i = 0; i < 8; ++i) {
                float v = c[s][i] + bi;
                v = v > 0.0f ? v : 0.0f;
                H[(size_t)(rbase + i) * N1PAD + n] = f2bf(v);
            }
        } else {
            float* out = (float*)outv;
            if (n < DFEAT) {
                float bi = bias[n];
#pragma unroll
                for (int i = 0; i < 8; ++i) {
                    int r = rbase + i;
                    if (r < NNODES) out[(size_t)r * DFEAT + n] = c[s][i] + bi;
                }
            }
        }
    }
}

// ---------------------------------------------------------------------------
// 5) BatchNorm: biased batch stats per column, then normalize in place
// ---------------------------------------------------------------------------
__global__ __launch_bounds__(256)
void bn_stats_kernel(const float* __restrict__ out,
                     float* __restrict__ mean, float* __restrict__ invstd) {
    __shared__ float ssum[256];
    __shared__ float ssq[256];
    int c = blockIdx.x;
    float s = 0.0f, q = 0.0f;
    for (int r = threadIdx.x; r < NNODES; r += 256) {
        float v = out[(size_t)r * DFEAT + c];
        s += v; q += v * v;
    }
    ssum[threadIdx.x] = s; ssq[threadIdx.x] = q;
    __syncthreads();
    for (int off = 128; off > 0; off >>= 1) {
        if (threadIdx.x < off) {
            ssum[threadIdx.x] += ssum[threadIdx.x + off];
            ssq[threadIdx.x]  += ssq[threadIdx.x + off];
        }
        __syncthreads();
    }
    if (threadIdx.x == 0) {
        float m   = ssum[0] / (float)NNODES;
        float var = ssq[0] / (float)NNODES - m * m;
        mean[c]   = m;
        invstd[c] = rsqrtf(var + 1e-5f);
    }
}

__global__ void bn_apply_kernel(float* __restrict__ out,
                                const float* __restrict__ mean,
                                const float* __restrict__ invstd,
                                const float* __restrict__ gamma,
                                const float* __restrict__ beta) {
    size_t idx = (size_t)blockIdx.x * blockDim.x + threadIdx.x;
    if (idx >= (size_t)NNODES * DFEAT) return;
    int c = (int)(idx % DFEAT);
    out[idx] = (out[idx] - mean[c]) * invstd[c] * gamma[c] + beta[c];
}

// ---------------------------------------------------------------------------
// launch
// ---------------------------------------------------------------------------
extern "C" void kernel_launch(void* const* d_in, const int* in_sizes, int n_in,
                              void* d_out, int out_size, void* d_ws, size_t ws_size,
                              hipStream_t stream) {
    (void)in_sizes; (void)n_in; (void)out_size; (void)ws_size;

    const float* node_feats = (const float*)d_in[0];
    const int*   src        = (const int*)  d_in[1];
    const int*   dst        = (const int*)  d_in[2];
    const int*   ef0        = (const int*)  d_in[3];
    const int*   ef1        = (const int*)  d_in[4];
    const float* emb_w0     = (const float*)d_in[5];
    const float* emb_b0     = (const float*)d_in[6];
    const float* emb_w1     = (const float*)d_in[7];
    const float* emb_b1     = (const float*)d_in[8];
    const float* mlp_w1     = (const float*)d_in[9];
    const float* mlp_b1     = (const float*)d_in[10];
    const float* mlp_w2     = (const float*)d_in[11];
    const float* mlp_b2     = (const float*)d_in[12];
    const float* bn_gamma   = (const float*)d_in[13];
    const float* bn_beta    = (const float*)d_in[14];
    float* out = (float*)d_out;

    // workspace layout (all offsets 256B aligned)
    char* ws = (char*)d_ws;
    float*          agg  = (float*)ws;                            // MPAD*K1PAD*4  = 128,040,960
    unsigned short* a1   = (unsigned short*)(ws + 128040960);     // MPAD*K1PAD*2  =  64,020,480
    unsigned short* h1   = (unsigned short*)(ws + 192061440);     // MPAD*N1PAD*2  = 128,040,960
    unsigned short* w1p  = (unsigned short*)(ws + 320102400);     // KT1*NT1*512*2 =     409,600
    unsigned short* w2p  = (unsigned short*)(ws + 320512000);     // KT2*NT2*512*2 =     491,520
    float*          mean = (float*)(ws + 321003520);
    float*          istd = mean + DFEAT;

    const size_t agg_elems = (size_t)MPAD * K1PAD;
    zero_f32_kernel<<<(unsigned)((agg_elems + 255) / 256), 256, 0, stream>>>(agg, agg_elems);

    scatter_kernel<<<NEDGES, 128, 0, stream>>>(node_feats, src, dst, ef0, ef1,
                                               emb_w0, emb_b0, emb_w1, emb_b1, agg);

    cvt_bf16_kernel<<<(unsigned)((agg_elems + 255) / 256), 256, 0, stream>>>(agg, a1, agg_elems);

    pack_w_kernel<<<(KT1 * NT1 * 512 + 255) / 256, 256, 0, stream>>>(mlp_w1, w1p, KT1, NT1, DFEAT, DHID);
    pack_w_kernel<<<(KT2 * NT2 * 512 + 255) / 256, 256, 0, stream>>>(mlp_w2, w2p, KT2, NT2, DHID, DFEAT);

    gemm_kernel<KT1, K1PAD, true ><<<dim3(MTILES, N1PAD / 128), 256, 0, stream>>>(a1, w1p, mlp_b1, h1);
    gemm_kernel<KT2, K2PAD, false><<<dim3(MTILES, N2PAD / 128), 256, 0, stream>>>(h1, w2p, mlp_b2, out);

    bn_stats_kernel<<<DFEAT, 256, 0, stream>>>(out, mean, istd);

    const size_t out_elems = (size_t)NNODES * DFEAT;
    bn_apply_kernel<<<(unsigned)((out_elems + 255) / 256), 256, 0, stream>>>(out, mean, istd, bn_gamma, bn_beta);
}
